// TransformerBasedRandomizedStackedEnsemble_7198365188224
// MI455X (gfx1250) — matrix operand order repaired and checked by reading, not compiled, not run on hardware
//
#include <hip/hip_runtime.h>
#include <hip/hip_bf16.h>
#include <cstring>
#include <stdint.h>

#define NCOMP 32
#define DD    17
#define WAVES 8
#define SPW   8
#define EPSC  1e-5f
#define SRO   21   // f32 row stride (odd -> conflict-free per-lane row access)
#define HRO   35   // score/h1 f32 row stride

typedef _Float16 h16;
typedef __attribute__((ext_vector_type(16))) _Float16 v16h;
typedef __attribute__((ext_vector_type(8)))  float    v8f;

struct Fids { unsigned char f[NCOMP][4]; };
struct CompP {
  const float* emb[4];
  const float* W1; const float* W2; const float* Wc;
  int fid[4]; int esz[4]; int in_f; int pad_;
};
struct AllComp { CompP c[NCOMP]; };

// ---------------- WMMA helpers (ISA 7.12.2 layouts, wave32) ----------------
__device__ __forceinline__ v16h ld_A(const h16* __restrict__ A, int lane) {
  const h16* r = A + (lane & 15) * 32;
  int kb = (lane >> 4) << 3;
  v16h a;
#pragma unroll
  for (int i = 0; i < 8; ++i) a[i] = r[kb + i];
#pragma unroll
  for (int i = 0; i < 8; ++i) a[8 + i] = r[16 + kb + i];
  return a;
}
__device__ __forceinline__ v16h ld_B(const h16* __restrict__ B, int stride, int col0, int lane) {
  const h16* r = B + col0 + (lane & 15);
  int kb = (lane >> 4) << 3;
  v16h b;
#pragma unroll
  for (int i = 0; i < 8; ++i) b[i] = r[(kb + i) * stride];
#pragma unroll
  for (int i = 0; i < 8; ++i) b[8 + i] = r[(16 + kb + i) * stride];
  return b;
}
__device__ __forceinline__ v8f wmma_f16(v16h a, v16h b, v8f c) {
  return __builtin_amdgcn_wmma_f32_16x16x32_f16(false, a, false, b, (short)0, c, false, false);
}

// ---------------- kernel 0: zero histograms (must run every call) ----------
__global__ void k_zero(int* __restrict__ cnts) {
  for (int i = threadIdx.x; i < 528; i += 256) cnts[i] = 0;
}

// ---------------- kernel 1: bit-pack codes + pattern histograms ------------
__global__ __launch_bounds__(256) void k_stats(const int* __restrict__ x, Fids fids,
                                               int* __restrict__ code, int* __restrict__ cnts, int N) {
  __shared__ int lpat[NCOMP * 16];
  __shared__ int lone[16];
  const int t = threadIdx.x;
  for (int i = t; i < NCOMP * 16; i += 256) lpat[i] = 0;
  if (t < 16) lone[t] = 0;
  __syncthreads();
  for (int n = blockIdx.x * 256 + t; n < N; n += gridDim.x * 256) {
    int cv = 0;
#pragma unroll
    for (int b = 0; b < 14; ++b) {
      int bit = x[n * 14 + b] & 1;
      cv |= bit << b;
      unsigned long long m = __ballot(bit);
      if ((t & 31) == 0 && m) atomicAdd(&lone[b], (int)__popcll(m));
    }
    code[n] = cv;
#pragma unroll
    for (int s = 0; s < NCOMP; ++s) {
      int p = 0;
#pragma unroll
      for (int j = 0; j < 4; ++j) p |= ((cv >> fids.f[s][j]) & 1) << j;
      atomicAdd(&lpat[s * 16 + p], 1);
    }
  }
  __syncthreads();
  for (int i = t; i < NCOMP * 16; i += 256) if (lpat[i]) atomicAdd(&cnts[i], lpat[i]);
  if (t < 14 && lone[t]) atomicAdd(&cnts[512 + t], lone[t]);
}

// ---------------- kernel 2: per-component 16-pattern tables ----------------
__global__ __launch_bounds__(512) void k_tables(AllComp AC, const int* __restrict__ cnts,
                                                float* __restrict__ tab, int N) {
  __shared__ float raw[16 * 32];
  __shared__ float meanv[32], scalev[32];
  __shared__ float featb[16 * 16];
  const int i = blockIdx.x;
  const CompP& cp = AC.c[i];
  const int t = threadIdx.x;
  const int* cntpat = cnts;
  const int* cnt1 = cnts + 512;
  const float invN = 1.0f / (float)N;
  {
    int p = t & 15, j = t >> 4;  // j in 0..31
    const float* W1r = cp.W1 + j * cp.in_f;
    float acc = 0.0f; int c = 0;
    for (int jj = 0; jj < 4; ++jj) {
      int bit = (p >> jj) & 1;
      int e = cp.esz[jj];
      if (e > 0) {
        const float* er = cp.emb[jj] + bit * e;
        for (int k = 0; k < e; ++k) acc += er[k] * W1r[c++];
      } else {
        float mu = (float)cnt1[cp.fid[jj]] * invN;
        float var = mu - mu * mu;               // Bernoulli biased var
        float val = ((float)bit - mu) * rsqrtf(var + EPSC);
        acc += val * W1r[c++];
      }
    }
    raw[p * 32 + j] = tanhf(acc);
  }
  __syncthreads();
  if (t < 32) {
    int j = t;
    float m = 0.0f, s2 = 0.0f;
    for (int p = 0; p < 16; ++p) {
      float cnt = (float)cntpat[i * 16 + p];
      float v = raw[p * 32 + j];
      m += cnt * v; s2 += cnt * v * v;
    }
    m *= invN; s2 *= invN;
    meanv[j] = m; scalev[j] = rsqrtf(s2 - m * m + EPSC);
  }
  __syncthreads();
  if (t < 256) {
    int p = t & 15, k = t >> 4;  // k in 0..15
    const float* W2r = cp.W2 + k * 32;
    float acc = 0.0f;
    for (int j = 0; j < 32; ++j) acc += (raw[p * 32 + j] - meanv[j]) * scalev[j] * W2r[j];
    float f = tanhf(acc);
    featb[p * 16 + k] = f;
    tab[(i * 16 + p) * DD + 1 + k] = f;
  }
  __syncthreads();
  if (t < 16) {
    int p = t;
    float acc = 0.0f;
    for (int k = 0; k < 16; ++k) acc += featb[p * 16 + k] * cp.Wc[k];
    tab[(i * 16 + p) * DD + 0] = acc;
  }
}

// ---------------- kernel 3: scatter src [NC,N,D] from tables ---------------
__global__ __launch_bounds__(256) void k_src(const int* __restrict__ code, const float* __restrict__ tab,
                                             Fids fids, float* __restrict__ outp, int N) {
  __shared__ float st[NCOMP * 16 * DD];
  const int t = threadIdx.x;
  for (int i = t; i < NCOMP * 16 * DD; i += 256) st[i] = tab[i];
  __syncthreads();
  for (int n = blockIdx.x * 256 + t; n < N; n += gridDim.x * 256) {
    int cv = code[n];
#pragma unroll
    for (int s = 0; s < NCOMP; ++s) {
      int p = 0;
#pragma unroll
      for (int j = 0; j < 4; ++j) p |= ((cv >> fids.f[s][j]) & 1) << j;
      const float* r = &st[(s * 16 + p) * DD];
      float* o = outp + (size_t)N + ((size_t)s * (size_t)N + (size_t)n) * DD;
#pragma unroll
      for (int d = 0; d < DD; ++d) o[d] = r[d];
    }
  }
}

// ---------------- kernel 4: transformer (1 sample / wave, WMMA) ------------
__global__ __launch_bounds__(256) void k_transformer(
    const float* __restrict__ srcg, float* __restrict__ outg,
    const float* __restrict__ in_w, const float* __restrict__ in_b,
    const float* __restrict__ out_w, const float* __restrict__ out_b,
    const float* __restrict__ l1_w, const float* __restrict__ l1_b,
    const float* __restrict__ l2_w, const float* __restrict__ l2_b,
    const float* __restrict__ n1_g, const float* __restrict__ n1_b,
    const float* __restrict__ n2_g, const float* __restrict__ n2_b,
    const float* __restrict__ ln_g, const float* __restrict__ ln_b,
    const float* __restrict__ cls_w, const float* __restrict__ cls_b, int N) {
  __shared__ h16 s_inwT[32 * 64];
  __shared__ h16 s_outwT[32 * 32], s_l1T[32 * 32], s_l2T[32 * 32];
  __shared__ float s_inb[64];
  __shared__ float s_outb[20], s_l1b[20], s_l2b[20];
  __shared__ float s_n1g[20], s_n1b[20], s_n2g[20], s_n2b[20], s_lng[20], s_lnb[20];
  __shared__ float s_cls[NCOMP * DD];
  __shared__ h16 A16a[WAVES][32 * 32];      // src -> exp(scores) -> h1
  __shared__ h16 A16b[WAVES][32 * 32];      // q -> o -> relu1
  __shared__ h16 B16a[WAVES][32 * 32];      // k^T (rows >= DD stay zero)
  __shared__ h16 B16b[WAVES][32 * 32];      // v   (cols >= DD stay zero)
  __shared__ float Sf32[WAVES][32 * SRO];   // src f32 residual
  __shared__ float Tf32[WAVES][32 * SRO];   // f32 GEMM outputs
  __shared__ float H1f[WAVES][32 * HRO];    // scores f32 / h1 f32
  __shared__ float rowinv[WAVES][32];
  __shared__ float ftrash[WAVES][32];
  __shared__ h16   htrash[WAVES][32];

  const int tid = threadIdx.x;
  const int wave = tid >> 5;
  const int lane = tid & 31;

  for (int i = tid; i < 32 * 64; i += 256) {
    int c = i >> 6, d = i & 63;
    s_inwT[i] = (c < DD && d < 3 * DD) ? (h16)in_w[d * DD + c] : (h16)0.0f;
  }
  for (int i = tid; i < 32 * 32; i += 256) {
    int c = i >> 5, d = i & 31;
    bool ok = (c < DD && d < DD);
    s_outwT[i] = ok ? (h16)out_w[d * DD + c] : (h16)0.0f;
    s_l1T[i]   = ok ? (h16)l1_w[d * DD + c]  : (h16)0.0f;
    s_l2T[i]   = ok ? (h16)l2_w[d * DD + c]  : (h16)0.0f;
  }
  if (tid < 64) s_inb[tid] = (tid < 3 * DD) ? in_b[tid] : 0.0f;
  if (tid < DD) {
    s_outb[tid] = out_b[tid]; s_l1b[tid] = l1_b[tid]; s_l2b[tid] = l2_b[tid];
    s_n1g[tid] = n1_g[tid]; s_n1b[tid] = n1_b[tid];
    s_n2g[tid] = n2_g[tid]; s_n2b[tid] = n2_b[tid];
    s_lng[tid] = ln_g[tid]; s_lnb[tid] = ln_b[tid];
  }
  for (int i = tid; i < NCOMP * DD; i += 256) s_cls[i] = cls_w[i];
  for (int i = lane; i < 32 * 32; i += 32) {            // establish zero padding once
    A16b[wave][i] = (h16)0.0f; B16a[wave][i] = (h16)0.0f; B16b[wave][i] = (h16)0.0f;
  }
  __syncthreads();

  const float qscale = rsqrtf((float)DD);

  for (int it = 0; it < SPW; ++it) {
    int n = blockIdx.x * (WAVES * SPW) + it * WAVES + wave;
    if (n >= N) n = N - 1;

    // 1) load src row (lane = component): f16 K-padded + f32 residual copy
    {
      const float* sr = srcg + ((size_t)lane * (size_t)N + (size_t)n) * DD;
      __builtin_prefetch(sr + (size_t)WAVES * DD, 0, 1);   // next iteration's row
      float* sf = &Sf32[wave][lane * SRO];
      h16* ar = &A16a[wave][lane * 32];
#pragma unroll
      for (int d = 0; d < DD; ++d) { float v = sr[d]; sf[d] = v; ar[d] = (h16)v; }
#pragma unroll
      for (int d = DD; d < 32; ++d) ar[d] = (h16)0.0f;
    }
    __syncthreads();

    // 2) qkv = src @ in_w^T + in_b  (branchless routing: q->A16b, k^T->B16a, v->B16b)
#pragma unroll
    for (int mt = 0; mt < 2; ++mt) {
      v16h a = ld_A(&A16a[wave][mt * 16 * 32], lane);
#pragma unroll
      for (int nt = 0; nt < 4; ++nt) {
        v8f c = {};
        c = wmma_f16(a, ld_B(s_inwT, 64, nt * 16, lane), c);
        int dd = nt * 16 + (lane & 15);
        int mB = mt * 16 + ((lane >> 4) << 3);
        float bia = s_inb[dd];
        h16* base; int strd; float scl;
        if (dd < DD)          { base = &A16b[wave][mB * 32 + dd];           strd = 32; scl = qscale; }
        else if (dd < 2 * DD) { base = &B16a[wave][(dd - DD) * 32 + mB];    strd = 1;  scl = 1.0f; }
        else if (dd < 3 * DD) { base = &B16b[wave][mB * 32 + (dd - 2*DD)];  strd = 32; scl = 1.0f; }
        else                  { base = &htrash[wave][lane];                 strd = 0;  scl = 0.0f; }
#pragma unroll
        for (int r = 0; r < 8; ++r) base[r * strd] = (h16)((c[r] + bia) * scl);
      }
    }
    __syncthreads();

    // 3) scores = q @ k^T -> H1f (f32); softmax per row (lane = row):
    //    unnormalized exp -> A16a (f16), 1/rowsum -> rowinv
#pragma unroll
    for (int mt = 0; mt < 2; ++mt) {
      v16h a = ld_A(&A16b[wave][mt * 16 * 32], lane);
#pragma unroll
      for (int nt = 0; nt < 2; ++nt) {
        v8f c = {};
        c = wmma_f16(a, ld_B(&B16a[wave][0], 32, nt * 16, lane), c);
        int tcol = nt * 16 + (lane & 15);
        int mB = mt * 16 + ((lane >> 4) << 3);
#pragma unroll
        for (int r = 0; r < 8; ++r) H1f[wave][(mB + r) * HRO + tcol] = c[r];
      }
    }
    __syncthreads();
    {
      const float* rp = &H1f[wave][lane * HRO];
      float mx = rp[0];
      for (int tt = 1; tt < 32; ++tt) mx = fmaxf(mx, rp[tt]);
      h16* pr = &A16a[wave][lane * 32];
      float sum = 0.0f;
      for (int tt = 0; tt < 32; ++tt) { float e = expf(rp[tt] - mx); sum += e; pr[tt] = (h16)e; }
      rowinv[wave][lane] = 1.0f / sum;
    }
    __syncthreads();

    // 4) o = softmax(P) @ v -> A16b  (1/rowsum folded into the store)
#pragma unroll
    for (int mt = 0; mt < 2; ++mt) {
      v16h a = ld_A(&A16a[wave][mt * 16 * 32], lane);
#pragma unroll
      for (int nt = 0; nt < 2; ++nt) {
        v8f c = {};
        c = wmma_f16(a, ld_B(&B16b[wave][0], 32, nt * 16, lane), c);
        int d = nt * 16 + (lane & 15);
        int mB = mt * 16 + ((lane >> 4) << 3);
#pragma unroll
        for (int r = 0; r < 8; ++r)
          A16b[wave][(mB + r) * 32 + d] = (h16)(c[r] * rowinv[wave][mB + r]);
      }
    }
    __syncthreads();

    // 5) o2 = o @ out_w^T + out_b -> Tf32 (branchless trash for d >= DD)
#pragma unroll
    for (int mt = 0; mt < 2; ++mt) {
      v16h a = ld_A(&A16b[wave][mt * 16 * 32], lane);
#pragma unroll
      for (int nt = 0; nt < 2; ++nt) {
        v8f c = {};
        c = wmma_f16(a, ld_B(s_outwT, 32, nt * 16, lane), c);
        int d = nt * 16 + (lane & 15);
        int mB = mt * 16 + ((lane >> 4) << 3);
        float* base; int strd; float bia;
        if (d < DD) { base = &Tf32[wave][mB * SRO + d]; strd = SRO; bia = s_outb[d]; }
        else        { base = &ftrash[wave][lane];       strd = 0;   bia = 0.0f; }
#pragma unroll
        for (int r = 0; r < 8; ++r) base[r * strd] = c[r] + bia;
      }
    }
    __syncthreads();

    // 6) h1 = LN(src + o2; n1) -> H1f (f32) and A16a (f16, zero-padded)
    {
      const float* sf = &Sf32[wave][lane * SRO];
      const float* tf = &Tf32[wave][lane * SRO];
      float v[DD]; float m = 0.0f;
#pragma unroll
      for (int d = 0; d < DD; ++d) { v[d] = sf[d] + tf[d]; m += v[d]; }
      m /= (float)DD;
      float var = 0.0f;
#pragma unroll
      for (int d = 0; d < DD; ++d) { float t = v[d] - m; var += t * t; }
      float inv = rsqrtf(var / (float)DD + EPSC);
      float* hb = &H1f[wave][lane * HRO];
      h16* ar = &A16a[wave][lane * 32];
#pragma unroll
      for (int d = 0; d < DD; ++d) { float h = (v[d] - m) * inv * s_n1g[d] + s_n1b[d]; hb[d] = h; ar[d] = (h16)h; }
#pragma unroll
      for (int d = DD; d < 32; ++d) ar[d] = (h16)0.0f;
    }
    __syncthreads();

    // 7) relu(h1 @ l1^T + l1_b) -> A16b (select, full write keeps padding)
#pragma unroll
    for (int mt = 0; mt < 2; ++mt) {
      v16h a = ld_A(&A16a[wave][mt * 16 * 32], lane);
#pragma unroll
      for (int nt = 0; nt < 2; ++nt) {
        v8f c = {};
        c = wmma_f16(a, ld_B(s_l1T, 32, nt * 16, lane), c);
        int d = nt * 16 + (lane & 15);
        int mB = mt * 16 + ((lane >> 4) << 3);
        float bia = (d < DD) ? s_l1b[d] : 0.0f;
        float msk = (d < DD) ? 1.0f : 0.0f;
#pragma unroll
        for (int r = 0; r < 8; ++r) {
          float vv = fmaxf(c[r] + bia, 0.0f) * msk;
          A16b[wave][(mB + r) * 32 + d] = (h16)vv;
        }
      }
    }
    __syncthreads();

    // 8) ff = relu1 @ l2^T + l2_b -> Tf32
#pragma unroll
    for (int mt = 0; mt < 2; ++mt) {
      v16h a = ld_A(&A16b[wave][mt * 16 * 32], lane);
#pragma unroll
      for (int nt = 0; nt < 2; ++nt) {
        v8f c = {};
        c = wmma_f16(a, ld_B(s_l2T, 32, nt * 16, lane), c);
        int d = nt * 16 + (lane & 15);
        int mB = mt * 16 + ((lane >> 4) << 3);
        float* base; int strd; float bia;
        if (d < DD) { base = &Tf32[wave][mB * SRO + d]; strd = SRO; bia = s_l2b[d]; }
        else        { base = &ftrash[wave][lane];       strd = 0;   bia = 0.0f; }
#pragma unroll
        for (int r = 0; r < 8; ++r) base[r * strd] = c[r] + bia;
      }
    }
    __syncthreads();

    // 9) h2 = LN(h1 + ff; n2); values = LN(h2; ln); classifier dot + wave reduce
    {
      const float* hb = &H1f[wave][lane * HRO];
      const float* tf = &Tf32[wave][lane * SRO];
      float v[DD]; float m = 0.0f;
#pragma unroll
      for (int d = 0; d < DD; ++d) { v[d] = hb[d] + tf[d]; m += v[d]; }
      m /= (float)DD;
      float var = 0.0f;
#pragma unroll
      for (int d = 0; d < DD; ++d) { float t = v[d] - m; var += t * t; }
      float inv = rsqrtf(var / (float)DD + EPSC);
#pragma unroll
      for (int d = 0; d < DD; ++d) v[d] = (v[d] - m) * inv * s_n2g[d] + s_n2b[d];
      float m2 = 0.0f;
#pragma unroll
      for (int d = 0; d < DD; ++d) m2 += v[d];
      m2 /= (float)DD;
      float var2 = 0.0f;
#pragma unroll
      for (int d = 0; d < DD; ++d) { float t = v[d] - m2; var2 += t * t; }
      float inv2 = rsqrtf(var2 / (float)DD + EPSC);
      float partial = 0.0f;
#pragma unroll
      for (int d = 0; d < DD; ++d) {
        float val = (v[d] - m2) * inv2 * s_lng[d] + s_lnb[d];
        partial += val * s_cls[d * NCOMP + lane];
      }
#pragma unroll
      for (int off = 16; off > 0; off >>= 1) partial += __shfl_xor(partial, off, 32);
      if (lane == 0) outg[n] = partial + cls_b[0];
    }
  }
}

// ---------------- host: replicate np.random.RandomState(42).randint -------
static void compute_fidx(unsigned char F[NCOMP][4]) {
  uint32_t mt[624]; int mti;
  mt[0] = 42u;
  for (int i = 1; i < 624; ++i) mt[i] = 1812433253u * (mt[i - 1] ^ (mt[i - 1] >> 30)) + (uint32_t)i;
  mti = 624;
  auto next32 = [&]() -> uint32_t {
    if (mti >= 624) {
      for (int i = 0; i < 624; ++i) {
        uint32_t y = (mt[i] & 0x80000000u) | (mt[(i + 1) % 624] & 0x7fffffffu);
        mt[i] = mt[(i + 397) % 624] ^ (y >> 1) ^ ((y & 1u) ? 2567483615u : 0u);
      }
      mti = 0;
    }
    uint32_t y = mt[mti++];
    y ^= y >> 11; y ^= (y << 7) & 2636928640u; y ^= (y << 15) & 4022730752u; y ^= y >> 18;
    return y;
  };
  for (int i = 0; i < NCOMP; ++i)
    for (int j = 0; j < 4; ++j) {
      uint32_t v;
      do { v = next32() & 15u; } while (v > 13u);
      F[i][j] = (unsigned char)v;
    }
}

extern "C" void kernel_launch(void* const* d_in, const int* in_sizes, int n_in,
                              void* d_out, int out_size, void* d_ws, size_t ws_size,
                              hipStream_t stream) {
  static const int EMBT[14] = {-1, 5, -1, 8, -1, 4, 8, 3, 3, 1, -1, -1, -1, 21};
  unsigned char F[NCOMP][4];
  compute_fidx(F);
  Fids fids;
  memcpy(fids.f, F, sizeof(F));

  int idx = 0;
  const int* x = (const int*)d_in[idx++];
  const int N = in_sizes[0] / 14;

  AllComp AC;
  for (int i = 0; i < NCOMP; ++i) {
    CompP& c = AC.c[i];
    int inf = 0;
    for (int j = 0; j < 4; ++j) {
      int fid = F[i][j];
      c.fid[j] = fid;
      int e = EMBT[fid];
      if (e > 0) { c.emb[j] = (const float*)d_in[idx++]; c.esz[j] = e; inf += e; }
      else       { c.emb[j] = nullptr;                   c.esz[j] = 0; inf += 1; }
    }
    c.W1 = (const float*)d_in[idx++];
    c.W2 = (const float*)d_in[idx++];
    c.Wc = (const float*)d_in[idx++];
    c.in_f = inf; c.pad_ = 0;
  }
  const float* in_w  = (const float*)d_in[idx++];
  const float* in_b  = (const float*)d_in[idx++];
  const float* out_w = (const float*)d_in[idx++];
  const float* out_b = (const float*)d_in[idx++];
  const float* l1_w  = (const float*)d_in[idx++];
  const float* l1_b  = (const float*)d_in[idx++];
  const float* l2_w  = (const float*)d_in[idx++];
  const float* l2_b  = (const float*)d_in[idx++];
  const float* n1_g  = (const float*)d_in[idx++];
  const float* n1_b  = (const float*)d_in[idx++];
  const float* n2_g  = (const float*)d_in[idx++];
  const float* n2_b  = (const float*)d_in[idx++];
  const float* ln_g  = (const float*)d_in[idx++];
  const float* ln_b  = (const float*)d_in[idx++];
  const float* cls_w = (const float*)d_in[idx++];
  const float* cls_b = (const float*)d_in[idx++];

  int* code = (int*)d_ws;
  int* cnts = code + N;                 // 512 pattern counts + 14 ones counts
  float* tab = (float*)(cnts + 528);    // [32][16][17] f32
  float* out = (float*)d_out;           // [N] logits, then [32][N][17] src

  k_zero<<<1, 256, 0, stream>>>(cnts);
  k_stats<<<(N + 255) / 256, 256, 0, stream>>>(x, fids, code, cnts, N);
  k_tables<<<NCOMP, 512, 0, stream>>>(AC, cnts, tab, N);
  k_src<<<512, 256, 0, stream>>>(code, tab, fids, out, N);
  int tb = (N + WAVES * SPW - 1) / (WAVES * SPW);
  k_transformer<<<tb, 256, 0, stream>>>(out + N, out,
                                        in_w, in_b, out_w, out_b, l1_w, l1_b, l2_w, l2_b,
                                        n1_g, n1_b, n2_g, n2_b, ln_g, ln_b, cls_w, cls_b, N);
}
